// WanAttentionBlock_71957882077556
// MI455X (gfx1250) — matrix operand order repaired and checked by reading, compiled, not run on hardware
//
#include <hip/hip_runtime.h>
#include <hip/hip_bf16.h>

typedef __attribute__((ext_vector_type(16))) __bf16 bf16x16;
typedef __attribute__((ext_vector_type(8)))  __bf16 bf16x8;
typedef __attribute__((ext_vector_type(8)))  float  f32x8;

#define DIM 1536
#define FFN_DIM 8960
#define NH 12
#define HD 128
#define SEQ 3200
#define SEQ2 512

// ---------------------------------------------------------------------------
// em = modulation + e   (6*1536)
__global__ void addmod_kernel(const float* __restrict__ mod, const float* __restrict__ e,
                              float* __restrict__ em) {
    int i = blockIdx.x * 256 + threadIdx.x;
    if (i < 6 * DIM) em[i] = mod[i] + e[i];
}

// ---------------------------------------------------------------------------
// fp32 [K][N] -> bf16 [N][K]  (tiled transpose + convert)
__global__ __launch_bounds__(256) void wtrans_kernel(const float* __restrict__ in,
                                                     __bf16* __restrict__ out,
                                                     int K, int N) {
    __shared__ float t[32][33];
    int n0 = blockIdx.x * 32, k0 = blockIdx.y * 32;
    int lx = threadIdx.x & 31, ly = threadIdx.x >> 5;   // 32 x 8
    for (int i = 0; i < 4; ++i)
        t[ly + 8 * i][lx] = in[(size_t)(k0 + ly + 8 * i) * N + n0 + lx];
    __syncthreads();
    for (int i = 0; i < 4; ++i)
        out[(size_t)(n0 + ly + 8 * i) * K + k0 + lx] = (__bf16)t[lx][ly + 8 * i];
}

// ---------------------------------------------------------------------------
// fp32 -> bf16 elementwise
__global__ void cvt_bf16_kernel(const float* __restrict__ in, __bf16* __restrict__ out, size_t n) {
    size_t i = (size_t)blockIdx.x * 256 + threadIdx.x;
    if (i < n) out[i] = (__bf16)in[i];
}

// ---------------------------------------------------------------------------
// LayerNorm (+optional affine) (+optional adaLN shift/scale), bf16 out. One block per row.
__global__ __launch_bounds__(256) void ln_mod_kernel(const float* __restrict__ x,
                                                     const float* __restrict__ gamma,
                                                     const float* __restrict__ beta,
                                                     const float* __restrict__ shift,
                                                     const float* __restrict__ scale1p,
                                                     __bf16* __restrict__ out, int ncols) {
    int row = blockIdx.x, tid = threadIdx.x;
    const float* xr = x + (size_t)row * ncols;
    float s = 0.f, s2 = 0.f;
    for (int c = tid; c < ncols; c += 256) { float v = xr[c]; s += v; s2 += v * v; }
    __shared__ float rs[256], rq[256];
    rs[tid] = s; rq[tid] = s2; __syncthreads();
    for (int o = 128; o > 0; o >>= 1) {
        if (tid < o) { rs[tid] += rs[tid + o]; rq[tid] += rq[tid + o]; }
        __syncthreads();
    }
    float mu = rs[0] / ncols;
    float var = rq[0] / ncols - mu * mu;
    float inv = rsqrtf(var + 1e-6f);
    for (int c = tid; c < ncols; c += 256) {
        float v = (xr[c] - mu) * inv;
        if (gamma)   v = v * gamma[c] + beta[c];
        if (scale1p) v = v * (1.f + scale1p[c]) + shift[c];
        out[(size_t)row * ncols + c] = (__bf16)v;
    }
}

// ---------------------------------------------------------------------------
// WMMA GEMM: C[M,N](f32,+bias) = A[M,K](bf16) @ Bt[N,K](bf16)^T
// 128x128 block tile, 8 waves (4 row x 2 col), 32x64 per wave = 2x4 WMMA tiles,
// double-buffered LDS, one barrier per K-step. All fragments loaded before the
// WMMA block so the 8 WMMAs issue back-to-back (no ds-wait fences between them).
#define TM 128
#define TN 128
#define TK 32
__global__ __launch_bounds__(256) void gemm_bf16_kernel(const __bf16* __restrict__ A,
                                                        const __bf16* __restrict__ Bt,
                                                        const float* __restrict__ bias,
                                                        float* __restrict__ C,
                                                        int M, int N, int K) {
    __shared__ __bf16 sA[2][TM][TK + 8];
    __shared__ __bf16 sB[2][TN][TK + 8];
    const int tid = threadIdx.x;
    const int lane = tid & 31, wave = tid >> 5;
    const int wr = wave & 3;        // 4 wave-rows of 32
    const int wc = wave >> 2;       // 2 wave-cols of 64
    const int hi = lane >> 4, l15 = lane & 15;
    const int m0 = blockIdx.y * TM;
    const int n0 = blockIdx.x * TN;
    const int lrow = tid >> 1;          // 0..127
    const int lcol = (tid & 1) * 16;    // 0,16

    const __bf16* pa = &A [(size_t)(m0 + lrow) * K + lcol];
    const __bf16* pb = &Bt[(size_t)(n0 + lrow) * K + lcol];

    f32x8 acc[2][4];
    for (int i = 0; i < 2; ++i) for (int j = 0; j < 4; ++j) acc[i][j] = (f32x8)(0.0f);

    // prologue: stage K-step 0
    {
        bf16x8 ra0 = *(const bf16x8*)pa, ra1 = *(const bf16x8*)(pa + 8);
        bf16x8 rb0 = *(const bf16x8*)pb, rb1 = *(const bf16x8*)(pb + 8);
        pa += TK; pb += TK;
        *(bf16x8*)&sA[0][lrow][lcol]     = ra0;
        *(bf16x8*)&sA[0][lrow][lcol + 8] = ra1;
        *(bf16x8*)&sB[0][lrow][lcol]     = rb0;
        *(bf16x8*)&sB[0][lrow][lcol + 8] = rb1;
    }

    const int nsteps = K / TK;
    for (int s = 0; s < nsteps; ++s) {
        const int cur = s & 1;
        __syncthreads();
        const bool pre = (s + 1 < nsteps);
        bf16x8 ra0, ra1, rb0, rb1;
        if (pre) {   // issue next tile's global loads early; latency hides under WMMA
            ra0 = *(const bf16x8*)pa; ra1 = *(const bf16x8*)(pa + 8);
            rb0 = *(const bf16x8*)pb; rb1 = *(const bf16x8*)(pb + 8);
            pa += TK; pb += TK;
        }

        // load ALL fragments first (distinct registers), then back-to-back WMMA
        bf16x16 af[2], bfr[4];
        for (int i = 0; i < 2; ++i) {
            const __bf16* pr = &sA[cur][wr * 32 + i * 16 + l15][0];
            bf16x8 lo = *(const bf16x8*)&pr[hi * 8];
            bf16x8 h8 = *(const bf16x8*)&pr[16 + hi * 8];
            for (int t = 0; t < 8; ++t) { af[i][t] = lo[t]; af[i][t + 8] = h8[t]; }
        }
        for (int j = 0; j < 4; ++j) {
            const __bf16* pr = &sB[cur][wc * 64 + j * 16 + l15][0];
            bf16x8 lo = *(const bf16x8*)&pr[hi * 8];
            bf16x8 h8 = *(const bf16x8*)&pr[16 + hi * 8];
            for (int t = 0; t < 8; ++t) { bfr[j][t] = lo[t]; bfr[j][t + 8] = h8[t]; }
        }
        for (int j = 0; j < 4; ++j) {
            acc[0][j] = __builtin_amdgcn_wmma_f32_16x16x32_bf16(
                false, af[0], false, bfr[j], (short)0, acc[0][j], false, false);
            acc[1][j] = __builtin_amdgcn_wmma_f32_16x16x32_bf16(
                false, af[1], false, bfr[j], (short)0, acc[1][j], false, false);
        }

        if (pre) {   // fill the other buffer
            const int nxt = cur ^ 1;
            *(bf16x8*)&sA[nxt][lrow][lcol]     = ra0;
            *(bf16x8*)&sA[nxt][lrow][lcol + 8] = ra1;
            *(bf16x8*)&sB[nxt][lrow][lcol]     = rb0;
            *(bf16x8*)&sB[nxt][lrow][lcol + 8] = rb1;
        }
    }

    for (int i = 0; i < 2; ++i)
        for (int j = 0; j < 4; ++j) {
            int col = n0 + wc * 64 + j * 16 + l15;
            float bv = bias ? bias[col] : 0.f;
            for (int r = 0; r < 8; ++r) {
                int row = m0 + wr * 32 + i * 16 + hi * 8 + r;
                C[(size_t)row * N + col] = acc[i][j][r] + bv;
            }
        }
}

// ---------------------------------------------------------------------------
// RMS-norm (over 1536) + optional Wan-RoPE + pack bf16 to [NH][L][HD]. Block per row.
__global__ __launch_bounds__(256) void rms_rope_pack_kernel(const float* __restrict__ in,
                                                            const float* __restrict__ nw,
                                                            const float* __restrict__ cosT,
                                                            const float* __restrict__ sinT,
                                                            const int* __restrict__ gh,
                                                            const int* __restrict__ gw,
                                                            __bf16* __restrict__ out,
                                                            int L, int do_rope) {
    int s = blockIdx.x, tid = threadIdx.x;
    const float* row = in + (size_t)s * DIM;
    float ss = 0.f;
    for (int c = tid; c < DIM; c += 256) { float v = row[c]; ss += v * v; }
    __shared__ float red[256];
    red[tid] = ss; __syncthreads();
    for (int o = 128; o > 0; o >>= 1) { if (tid < o) red[tid] += red[tid + o]; __syncthreads(); }
    float inv = rsqrtf(red[0] / (float)DIM + 1e-6f);
    int fi = 0, hh = 0, ww = 0;
    if (do_rope) { int H = *gh, W = *gw; fi = s / (H * W); hh = (s / W) % H; ww = s % W; }
    for (int p = tid; p < NH * 64; p += 256) {   // 64 complex pairs per head
        int n = p >> 6, j = p & 63;              // c0=22, c1=21: [t|h|w]
        int c0 = n * HD + 2 * j;
        float re = row[c0] * inv * nw[c0];
        float im = row[c0 + 1] * inv * nw[c0 + 1];
        if (do_rope) {
            int trow = (j < 22) ? fi : ((j < 43) ? hh : ww);
            float cv = cosT[trow * 64 + j], sv = sinT[trow * 64 + j];
            float r2 = re * cv - im * sv;
            im = re * sv + im * cv;
            re = r2;
        }
        size_t o = ((size_t)n * L + s) * HD + 2 * j;
        out[o] = (__bf16)re; out[o + 1] = (__bf16)im;
    }
}

// ---------------------------------------------------------------------------
// f32 [L][1536] -> bf16 V^T [NH][HD][L]
__global__ void pack_vT_kernel(const float* __restrict__ in, __bf16* __restrict__ out, int L) {
    size_t i = (size_t)blockIdx.x * 256 + threadIdx.x;
    if (i >= (size_t)L * DIM) return;
    int s = (int)(i / DIM), c = (int)(i % DIM);
    int n = c >> 7, d = c & 127;
    out[((size_t)n * HD + d) * L + s] = (__bf16)in[i];
}

// ---------------------------------------------------------------------------
// Flash attention: Q [NH][Lq][128], K [NH][Lk][128], V^T [NH][128][Lk] -> out bf16 [Lq][1536]
__global__ __launch_bounds__(256) void attn_kernel(const __bf16* __restrict__ Q,
                                                   const __bf16* __restrict__ Kb,
                                                   const __bf16* __restrict__ VT,
                                                   __bf16* __restrict__ out,
                                                   int Lq, int Lk) {
    __shared__ __bf16 sQ[64][HD + 8];
    __shared__ __bf16 sK[64][HD + 8];
    __shared__ __bf16 sV[HD][64 + 8];
    __shared__ float  sS[64][64 + 4];
    __shared__ __bf16 sP[64][64 + 8];
    __shared__ float  mrow[64], lrow[64], crow[64];

    const int head = blockIdx.y;
    const int q0 = blockIdx.x * 64;
    const int tid = threadIdx.x, lane = tid & 31, wave = tid >> 5;
    const int qw = wave & 3;      // q strip of 16
    const int dw = wave >> 2;     // 0..1: 32-key halves / 64-dim halves
    const int hi = lane >> 4, l15 = lane & 15;
    const float sm_scale = 0.08838834764831845f;  // 128^-0.5

    const __bf16* Qh = Q  + (size_t)head * Lq * HD;
    const __bf16* Kh = Kb + (size_t)head * Lk * HD;
    const __bf16* Vh = VT + (size_t)head * HD * Lk;

    {   // Q tile once
        int r = tid >> 2, c = (tid & 3) * 32;
        for (int u = 0; u < 4; ++u)
            *(bf16x8*)&sQ[r][c + u * 8] = *(const bf16x8*)&Qh[(size_t)(q0 + r) * HD + c + u * 8];
    }
    if (tid < 64) { mrow[tid] = -1e30f; lrow[tid] = 0.f; }
    f32x8 oacc[4];
    for (int t = 0; t < 4; ++t) oacc[t] = (f32x8)(0.0f);
    __syncthreads();

    for (int k0 = 0; k0 < Lk; k0 += 64) {
        {   // K tile [64 keys][128], V^T tile [128 d][64 keys]
            int r = tid >> 2, c = (tid & 3) * 32;
            for (int u = 0; u < 4; ++u)
                *(bf16x8*)&sK[r][c + u * 8] = *(const bf16x8*)&Kh[(size_t)(k0 + r) * HD + c + u * 8];
            int vr = tid >> 1, vc = (tid & 1) * 32;
            for (int u = 0; u < 4; ++u)
                *(bf16x8*)&sV[vr][vc + u * 8] = *(const bf16x8*)&Vh[(size_t)vr * Lk + k0 + vc + u * 8];
        }
        __syncthreads();

        // S = Q @ K^T : wave -> q strip qw, keys [dw*32, dw*32+32)
        f32x8 sacc[2];
        sacc[0] = (f32x8)(0.0f); sacc[1] = (f32x8)(0.0f);
        for (int kk = 0; kk < HD; kk += 32) {
            bf16x16 aq, bk[2];
            { const __bf16* pr = &sQ[qw * 16 + l15][kk];
              bf16x8 lo = *(const bf16x8*)&pr[hi * 8];
              bf16x8 h8 = *(const bf16x8*)&pr[16 + hi * 8];
              for (int t = 0; t < 8; ++t) { aq[t] = lo[t]; aq[t + 8] = h8[t]; } }
            for (int j = 0; j < 2; ++j) {
                const __bf16* pr = &sK[dw * 32 + j * 16 + l15][kk];
                bf16x8 lo = *(const bf16x8*)&pr[hi * 8];
                bf16x8 h8 = *(const bf16x8*)&pr[16 + hi * 8];
                for (int t = 0; t < 8; ++t) { bk[j][t] = lo[t]; bk[j][t + 8] = h8[t]; }
            }
            for (int j = 0; j < 2; ++j)
                sacc[j] = __builtin_amdgcn_wmma_f32_16x16x32_bf16(
                    false, aq, false, bk[j], (short)0, sacc[j], false, false);
        }
        for (int j = 0; j < 2; ++j)
            for (int r = 0; r < 8; ++r)
                sS[qw * 16 + hi * 8 + r][dw * 32 + j * 16 + l15] = sacc[j][r] * sm_scale;
        __syncthreads();

        // online softmax per row
        if (tid < 64) {
            int r = tid;
            float mx = -1e30f;
            for (int j = 0; j < 64; ++j) mx = fmaxf(mx, sS[r][j]);
            float mo = mrow[r], mn = fmaxf(mo, mx);
            float corr = __expf(mo - mn);
            float sum = 0.f;
            for (int j = 0; j < 64; ++j) {
                float pv = __expf(sS[r][j] - mn);
                sum += pv;
                sP[r][j] = (__bf16)pv;
            }
            lrow[r] = lrow[r] * corr + sum;
            mrow[r] = mn; crow[r] = corr;
        }
        __syncthreads();

        for (int t = 0; t < 4; ++t)
            for (int r = 0; r < 8; ++r)
                oacc[t][r] *= crow[qw * 16 + hi * 8 + r];

        // O += P @ V : wave -> q strip qw, dims [dw*64, dw*64+64)
        for (int kk = 0; kk < 64; kk += 32) {
            bf16x16 ap, bv[4];
            { const __bf16* pr = &sP[qw * 16 + l15][kk];
              bf16x8 lo = *(const bf16x8*)&pr[hi * 8];
              bf16x8 h8 = *(const bf16x8*)&pr[16 + hi * 8];
              for (int t = 0; t < 8; ++t) { ap[t] = lo[t]; ap[t + 8] = h8[t]; } }
            for (int t = 0; t < 4; ++t) {
                const __bf16* pr = &sV[dw * 64 + t * 16 + l15][kk];
                bf16x8 lo = *(const bf16x8*)&pr[hi * 8];
                bf16x8 h8 = *(const bf16x8*)&pr[16 + hi * 8];
                for (int u = 0; u < 8; ++u) { bv[t][u] = lo[u]; bv[t][u + 8] = h8[u]; }
            }
            for (int t = 0; t < 4; ++t)
                oacc[t] = __builtin_amdgcn_wmma_f32_16x16x32_bf16(
                    false, ap, false, bv[t], (short)0, oacc[t], false, false);
        }
        __syncthreads();
    }

    for (int t = 0; t < 4; ++t)
        for (int r = 0; r < 8; ++r) {
            int row = qw * 16 + hi * 8 + r;
            float o = oacc[t][r] / lrow[row];
            int d = dw * 64 + t * 16 + l15;
            out[(size_t)(q0 + row) * DIM + head * HD + d] = (__bf16)o;
        }
}

// ---------------------------------------------------------------------------
__global__ void gelu_pack_kernel(const float* __restrict__ in, __bf16* __restrict__ out, size_t n) {
    size_t i = (size_t)blockIdx.x * 256 + threadIdx.x;
    if (i >= n) return;
    float x = in[i];
    float t = tanhf(0.7978845608028654f * (x + 0.044715f * x * x * x));
    out[i] = (__bf16)(0.5f * x * (1.f + t));
}

// out = x + y * colscale (colscale may be null -> 1)
__global__ void residual_kernel(const float* __restrict__ x, const float* __restrict__ y,
                                const float* __restrict__ colscale, float* __restrict__ out,
                                int ncols, size_t n) {
    size_t i = (size_t)blockIdx.x * 256 + threadIdx.x;
    if (i >= n) return;
    int c = (int)(i % ncols);
    float s = colscale ? colscale[c] : 1.f;
    out[i] = x[i] + y[i] * s;
}

// ---------------------------------------------------------------------------
extern "C" void kernel_launch(void* const* d_in, const int* in_sizes, int n_in,
                              void* d_out, int out_size, void* d_ws, size_t ws_size,
                              hipStream_t stream) {
    const float* x       = (const float*)d_in[0];
    const float* e       = (const float*)d_in[1];
    const float* context = (const float*)d_in[2];
    const float* fcos    = (const float*)d_in[3];
    const float* fsin    = (const float*)d_in[4];
    const float* mod     = (const float*)d_in[5];
    const float* sa_qw = (const float*)d_in[6];  const float* sa_qb = (const float*)d_in[7];
    const float* sa_kw = (const float*)d_in[8];  const float* sa_kb = (const float*)d_in[9];
    const float* sa_vw = (const float*)d_in[10]; const float* sa_vb = (const float*)d_in[11];
    const float* sa_ow = (const float*)d_in[12]; const float* sa_ob = (const float*)d_in[13];
    const float* sa_nq = (const float*)d_in[14]; const float* sa_nk = (const float*)d_in[15];
    const float* ca_qw = (const float*)d_in[16]; const float* ca_qb = (const float*)d_in[17];
    const float* ca_kw = (const float*)d_in[18]; const float* ca_kb = (const float*)d_in[19];
    const float* ca_vw = (const float*)d_in[20]; const float* ca_vb = (const float*)d_in[21];
    const float* ca_ow = (const float*)d_in[22]; const float* ca_ob = (const float*)d_in[23];
    const float* ca_nq = (const float*)d_in[24]; const float* ca_nk = (const float*)d_in[25];
    const float* n3_w  = (const float*)d_in[26]; const float* n3_b  = (const float*)d_in[27];
    const float* ffn_w1 = (const float*)d_in[28]; const float* ffn_b1 = (const float*)d_in[29];
    const float* ffn_w2 = (const float*)d_in[30]; const float* ffn_b2 = (const float*)d_in[31];
    const int* gh = (const int*)d_in[33];
    const int* gw = (const int*)d_in[34];
    (void)in_sizes; (void)n_in; (void)out_size; (void)ws_size;

    char* ws = (char*)d_ws;
    size_t off = 0;
    auto alloc = [&](size_t bytes) -> char* {
        char* p = ws + off;
        off += (bytes + 255) & ~(size_t)255;
        return p;
    };
    const size_t W2 = (size_t)DIM * DIM;

    float*  em     = (float*) alloc(6 * DIM * 4);
    __bf16* saqT   = (__bf16*)alloc(W2 * 2);
    __bf16* sakT   = (__bf16*)alloc(W2 * 2);
    __bf16* savT   = (__bf16*)alloc(W2 * 2);
    __bf16* saoT   = (__bf16*)alloc(W2 * 2);
    __bf16* caqT   = (__bf16*)alloc(W2 * 2);
    __bf16* cakT   = (__bf16*)alloc(W2 * 2);
    __bf16* cavT   = (__bf16*)alloc(W2 * 2);
    __bf16* caoT   = (__bf16*)alloc(W2 * 2);
    __bf16* w1T    = (__bf16*)alloc((size_t)FFN_DIM * DIM * 2);
    __bf16* w2T    = (__bf16*)alloc((size_t)DIM * FFN_DIM * 2);
    __bf16* xs_bf  = (__bf16*)alloc((size_t)SEQ * DIM * 2);
    __bf16* ctx_bf = (__bf16*)alloc((size_t)SEQ2 * DIM * 2);
    __bf16* qh     = (__bf16*)alloc((size_t)NH * SEQ * HD * 2);
    __bf16* kh     = (__bf16*)alloc((size_t)NH * SEQ * HD * 2);
    __bf16* vT     = (__bf16*)alloc((size_t)NH * HD * SEQ * 2);
    __bf16* attnbf = (__bf16*)alloc((size_t)SEQ * DIM * 2);
    __bf16* h1     = (__bf16*)alloc((size_t)SEQ * FFN_DIM * 2);
    float*  x2     = (float*) alloc((size_t)SEQ * DIM * 4);
    float*  tmp    = (float*) alloc((size_t)SEQ * FFN_DIM * 4);

    auto gemm = [&](const __bf16* A, const __bf16* Bt, const float* bias, float* C,
                    int M, int N, int K) {
        gemm_bf16_kernel<<<dim3(N / TN, M / TM), 256, 0, stream>>>(A, Bt, bias, C, M, N, K);
    };
    auto blocks = [](size_t n) { return (unsigned)((n + 255) / 256); };

    // --- constants / weight prep ---
    addmod_kernel<<<blocks(6 * DIM), 256, 0, stream>>>(mod, e, em);
    wtrans_kernel<<<dim3(DIM / 32, DIM / 32), 256, 0, stream>>>(sa_qw, saqT, DIM, DIM);
    wtrans_kernel<<<dim3(DIM / 32, DIM / 32), 256, 0, stream>>>(sa_kw, sakT, DIM, DIM);
    wtrans_kernel<<<dim3(DIM / 32, DIM / 32), 256, 0, stream>>>(sa_vw, savT, DIM, DIM);
    wtrans_kernel<<<dim3(DIM / 32, DIM / 32), 256, 0, stream>>>(sa_ow, saoT, DIM, DIM);
    wtrans_kernel<<<dim3(DIM / 32, DIM / 32), 256, 0, stream>>>(ca_qw, caqT, DIM, DIM);
    wtrans_kernel<<<dim3(DIM / 32, DIM / 32), 256, 0, stream>>>(ca_kw, cakT, DIM, DIM);
    wtrans_kernel<<<dim3(DIM / 32, DIM / 32), 256, 0, stream>>>(ca_vw, cavT, DIM, DIM);
    wtrans_kernel<<<dim3(DIM / 32, DIM / 32), 256, 0, stream>>>(ca_ow, caoT, DIM, DIM);
    wtrans_kernel<<<dim3(FFN_DIM / 32, DIM / 32), 256, 0, stream>>>(ffn_w1, w1T, DIM, FFN_DIM);
    wtrans_kernel<<<dim3(DIM / 32, FFN_DIM / 32), 256, 0, stream>>>(ffn_w2, w2T, FFN_DIM, DIM);

    // --- self attention ---
    ln_mod_kernel<<<SEQ, 256, 0, stream>>>(x, (const float*)nullptr, (const float*)nullptr,
                                           em + 0 * DIM, em + 1 * DIM, xs_bf, DIM);
    gemm(xs_bf, saqT, sa_qb, tmp, SEQ, DIM, DIM);
    rms_rope_pack_kernel<<<SEQ, 256, 0, stream>>>(tmp, sa_nq, fcos, fsin, gh, gw, qh, SEQ, 1);
    gemm(xs_bf, sakT, sa_kb, tmp, SEQ, DIM, DIM);
    rms_rope_pack_kernel<<<SEQ, 256, 0, stream>>>(tmp, sa_nk, fcos, fsin, gh, gw, kh, SEQ, 1);
    gemm(xs_bf, savT, sa_vb, tmp, SEQ, DIM, DIM);
    pack_vT_kernel<<<blocks((size_t)SEQ * DIM), 256, 0, stream>>>(tmp, vT, SEQ);
    attn_kernel<<<dim3(SEQ / 64, NH), 256, 0, stream>>>(qh, kh, vT, attnbf, SEQ, SEQ);
    gemm(attnbf, saoT, sa_ob, tmp, SEQ, DIM, DIM);
    residual_kernel<<<blocks((size_t)SEQ * DIM), 256, 0, stream>>>(
        x, tmp, em + 2 * DIM, x2, DIM, (size_t)SEQ * DIM);

    // --- cross attention ---
    ln_mod_kernel<<<SEQ, 256, 0, stream>>>(x2, n3_w, n3_b,
                                           (const float*)nullptr, (const float*)nullptr, xs_bf, DIM);
    cvt_bf16_kernel<<<blocks((size_t)SEQ2 * DIM), 256, 0, stream>>>(context, ctx_bf, (size_t)SEQ2 * DIM);
    gemm(xs_bf, caqT, ca_qb, tmp, SEQ, DIM, DIM);
    rms_rope_pack_kernel<<<SEQ, 256, 0, stream>>>(tmp, ca_nq, fcos, fsin, gh, gw, qh, SEQ, 0);
    gemm(ctx_bf, cakT, ca_kb, tmp, SEQ2, DIM, DIM);
    rms_rope_pack_kernel<<<SEQ2, 256, 0, stream>>>(tmp, ca_nk, fcos, fsin, gh, gw, kh, SEQ2, 0);
    gemm(ctx_bf, cavT, ca_vb, tmp, SEQ2, DIM, DIM);
    pack_vT_kernel<<<blocks((size_t)SEQ2 * DIM), 256, 0, stream>>>(tmp, vT, SEQ2);
    attn_kernel<<<dim3(SEQ / 64, NH), 256, 0, stream>>>(qh, kh, vT, attnbf, SEQ, SEQ2);
    gemm(attnbf, caoT, ca_ob, tmp, SEQ, DIM, DIM);
    residual_kernel<<<blocks((size_t)SEQ * DIM), 256, 0, stream>>>(
        x2, tmp, (const float*)nullptr, x2, DIM, (size_t)SEQ * DIM);

    // --- FFN ---
    ln_mod_kernel<<<SEQ, 256, 0, stream>>>(x2, (const float*)nullptr, (const float*)nullptr,
                                           em + 3 * DIM, em + 4 * DIM, xs_bf, DIM);
    gemm(xs_bf, w1T, ffn_b1, tmp, SEQ, FFN_DIM, DIM);
    gelu_pack_kernel<<<blocks((size_t)SEQ * FFN_DIM), 256, 0, stream>>>(tmp, h1, (size_t)SEQ * FFN_DIM);
    gemm(h1, w2T, ffn_b2, tmp, SEQ, DIM, FFN_DIM);
    residual_kernel<<<blocks((size_t)SEQ * DIM), 256, 0, stream>>>(
        x2, tmp, em + 5 * DIM, (float*)d_out, DIM, (size_t)SEQ * DIM);
}
